// MultiHeadAttention_44693429682777
// MI455X (gfx1250) — compile-verified, matrix operand order checked by reading, not run on hardware
//
#include <hip/hip_runtime.h>
#include <hip/hip_bf16.h>
#include <math.h>

#define HEAD_NUM 12
#define EMB      768
#define HEAD     64
#define B_       8
#define L_       1024
#define ROWS     (B_ * L_)   // 8192

typedef unsigned short ushort_t;
typedef __attribute__((ext_vector_type(16))) __bf16        v16bf;
typedef __attribute__((ext_vector_type(8)))  float         v8f;
typedef __attribute__((ext_vector_type(4)))  unsigned int  u32x4;
typedef __attribute__((ext_vector_type(4)))  unsigned short u16x4;

union Frag16 { u32x4 q[2]; v16bf v; };
union Pack8  { u32x4 q;    ushort_t s[8]; };

__device__ __forceinline__ ushort_t f32_bf16(float f) {
    unsigned int u = __float_as_uint(f);
    u += 0x7FFFu + ((u >> 16) & 1u);          // round-to-nearest-even
    return (ushort_t)(u >> 16);
}

__device__ __forceinline__ v8f zero8() {
    v8f z;
#pragma unroll
    for (int i = 0; i < 8; ++i) z[i] = 0.0f;
    return z;
}

// Load one 16-bit A/B WMMA fragment: two contiguous 8-element (16B) runs.
__device__ __forceinline__ v16bf load_frag(const ushort_t* p0, const ushort_t* p1) {
    Frag16 f;
    f.q[0] = *(const u32x4*)p0;
    f.q[1] = *(const u32x4*)p1;
    return f.v;
}

__device__ __forceinline__ v8f wmma_bf16(v16bf a, v16bf b, v8f c) {
    return __builtin_amdgcn_wmma_f32_16x16x32_bf16(false, a, false, b, (short)0, c, false, false);
}

// ---------------- Kernel 0a: fp32 -> bf16 convert (emb) ----------------
__global__ void __launch_bounds__(256) cvt_bf16(const float* __restrict__ src,
                                                ushort_t* __restrict__ dst, int n) {
    int i = (blockIdx.x * blockDim.x + threadIdx.x) * 4;
    if (i + 3 < n) {
        float4 v = *(const float4*)(src + i);
        u16x4 o;
        o[0] = f32_bf16(v.x); o[1] = f32_bf16(v.y);
        o[2] = f32_bf16(v.z); o[3] = f32_bf16(v.w);
        *(u16x4*)(dst + i) = o;
    }
}

// ---------------- Kernel 0b: W[k][n] fp32 -> Wt[n][k] bf16 ----------------
__global__ void __launch_bounds__(256) xpose_w(const float* __restrict__ W,
                                               ushort_t* __restrict__ Wt) {
    int idx = blockIdx.x * blockDim.x + threadIdx.x;   // over 768*768
    int n = idx / EMB, k = idx % EMB;
    Wt[idx] = f32_bf16(W[k * EMB + n]);
}

// ---------------- Kernel 1: fused QKV projection + 2D RoPE ----------------
// One wave -> one (16-row, one-head=64-col) tile of Q, K, V (12 accumulators).
__global__ void __launch_bounds__(256) qkv_rope(
    const ushort_t* __restrict__ embb,
    const ushort_t* __restrict__ Wqt, const ushort_t* __restrict__ Wkt,
    const ushort_t* __restrict__ Wvt,
    const int* __restrict__ pos,
    ushort_t* __restrict__ Qr, ushort_t* __restrict__ Kr, ushort_t* __restrict__ Vt) {

    const int lane = threadIdx.x & 31;
    const int wave = threadIdx.x >> 5;
    const int gw   = blockIdx.x * 8 + wave;        // 0..6143
    const int rt   = gw / HEAD_NUM;                // 0..511 row tiles
    const int h    = gw % HEAD_NUM;
    const int row0 = rt * 16;
    const int b    = row0 >> 10;
    const int l0   = row0 & (L_ - 1);
    const int half = lane >> 4;
    const int m16  = lane & 15;

    v8f accQ[4], accK[4], accV[4];
#pragma unroll
    for (int t = 0; t < 4; ++t) { accQ[t] = zero8(); accK[t] = zero8(); accV[t] = zero8(); }

    const ushort_t* arow = embb + (size_t)(row0 + m16) * EMB;

#pragma unroll 1
    for (int ks = 0; ks < EMB / 32; ++ks) {
        const int k0 = ks * 32 + half * 8;
        v16bf a = load_frag(arow + k0, arow + k0 + 16);
#pragma unroll
        for (int t = 0; t < 4; ++t) {
            const size_t wro = (size_t)(h * 64 + t * 16 + m16) * EMB + k0;
            v16bf bq = load_frag(Wqt + wro, Wqt + wro + 16);
            accQ[t] = wmma_bf16(a, bq, accQ[t]);
            v16bf bk = load_frag(Wkt + wro, Wkt + wro + 16);
            accK[t] = wmma_bf16(a, bk, accK[t]);
            v16bf bv = load_frag(Wvt + wro, Wvt + wro + 16);
            accV[t] = wmma_bf16(a, bv, accV[t]);
        }
    }

    // 2D RoPE in-register: cols j and j+16 share a lane across tile pairs (0,1) and (2,3).
    // inv_freq[i] = 100^(-i/16), i == lane's N index (m16) for both halves of each pair.
    const float inv = __expf(-(float)m16 * 0.28782317f); // ln(100)/16
    const int bh = b * HEAD_NUM + h;

#pragma unroll
    for (int r = 0; r < 8; ++r) {
        const int l = l0 + half * 8 + r;
        const float px = (float)pos[2 * l + 0];
        const float py = (float)pos[2 * l + 1];
        float sx, cx, sy, cy;
        __sincosf(px * inv, &sx, &cx);
        __sincosf(py * inv, &sy, &cy);
        float q0 = accQ[0][r], q1 = accQ[1][r], q2 = accQ[2][r], q3 = accQ[3][r];
        accQ[0][r] = q0 * cx - q1 * sx;  accQ[1][r] = q1 * cx + q0 * sx;
        accQ[2][r] = q2 * cy - q3 * sy;  accQ[3][r] = q3 * cy + q2 * sy;
        float k0 = accK[0][r], k1 = accK[1][r], k2 = accK[2][r], k3 = accK[3][r];
        accK[0][r] = k0 * cx - k1 * sx;  accK[1][r] = k1 * cx + k0 * sx;
        accK[2][r] = k2 * cy - k3 * sy;  accK[3][r] = k3 * cy + k2 * sy;
    }

#pragma unroll
    for (int t = 0; t < 4; ++t) {
        // Q,K row-major [bh][l][64] (A-side operands later)
#pragma unroll
        for (int r = 0; r < 8; ++r) {
            const int l = l0 + half * 8 + r;
            const size_t o = ((size_t)bh * L_ + l) * HEAD + t * 16 + m16;
            Qr[o] = f32_bf16(accQ[t][r]);
            Kr[o] = f32_bf16(accK[t][r]);
        }
        // V transposed [bh][d][l] -> packed 16B store (B-side operand later)
        Pack8 pv;
#pragma unroll
        for (int r = 0; r < 8; ++r) pv.s[r] = f32_bf16(accV[t][r]);
        *(u32x4*)(Vt + ((size_t)bh * HEAD + t * 16 + m16) * L_ + l0 + half * 8) = pv.q;
    }
}

// ---------------- Kernel 2: flash attention per (b,h) ----------------
#define PROW 40   // padded LDS row stride (ushorts) -> conflict-free b128 reads
__global__ void __launch_bounds__(256) attn(
    const ushort_t* __restrict__ Qr, const ushort_t* __restrict__ Kr,
    const ushort_t* __restrict__ Vt, ushort_t* __restrict__ AO) {

    __shared__ ushort_t Pl[8 * 16 * PROW];

    const int lane = threadIdx.x & 31;
    const int wave = threadIdx.x >> 5;
    const int bh   = blockIdx.x >> 3;          // 0..95
    const int qc   = blockIdx.x & 7;
    const int b    = bh / HEAD_NUM, h = bh % HEAD_NUM;
    const int half = lane >> 4;
    const int m16  = lane & 15;
    const int q0row = qc * 128 + wave * 16;

    const ushort_t* Q = Qr + (size_t)bh * L_ * HEAD;
    const ushort_t* K = Kr + (size_t)bh * L_ * HEAD;
    const ushort_t* V = Vt + (size_t)bh * HEAD * L_;

    const ushort_t* qrow = Q + (size_t)(q0row + m16) * HEAD;
    v16bf aq0 = load_frag(qrow + half * 8,      qrow + half * 8 + 16);
    v16bf aq1 = load_frag(qrow + 32 + half * 8, qrow + 32 + half * 8 + 16);

    v8f O[4];
#pragma unroll
    for (int t = 0; t < 4; ++t) O[t] = zero8();
    float mrun[8], lrun[8];
#pragma unroll
    for (int r = 0; r < 8; ++r) { mrun[r] = -1e30f; lrun[r] = 0.0f; }

    ushort_t* myP = Pl + wave * 16 * PROW;

#pragma unroll 1
    for (int kb = 0; kb < L_ / 32; ++kb) {
        const int c0 = kb * 32;
        v8f s0 = zero8(), s1 = zero8();
        {
            const ushort_t* kr0 = K + (size_t)(c0 + m16) * HEAD;
            const ushort_t* kr1 = K + (size_t)(c0 + 16 + m16) * HEAD;
            v16bf b00 = load_frag(kr0 + half * 8,      kr0 + half * 8 + 16);
            s0 = wmma_bf16(aq0, b00, s0);
            v16bf b01 = load_frag(kr1 + half * 8,      kr1 + half * 8 + 16);
            s1 = wmma_bf16(aq0, b01, s1);
            v16bf b10 = load_frag(kr0 + 32 + half * 8, kr0 + 32 + half * 8 + 16);
            s0 = wmma_bf16(aq1, b10, s0);
            v16bf b11 = load_frag(kr1 + 32 + half * 8, kr1 + 32 + half * 8 + 16);
            s1 = wmma_bf16(aq1, b11, s1);
        }
        // Online softmax. Row M=r+8*half lives in the lane's 16-lane half-group,
        // so xor masks 1,2,4,8 reduce exactly over one row (wave32).
#pragma unroll
        for (int r = 0; r < 8; ++r) {
            float v0 = s0[r] * 0.125f, v1 = s1[r] * 0.125f;  // 1/sqrt(64)
            float mx = fmaxf(v0, v1);
#pragma unroll
            for (int d = 1; d < 16; d <<= 1) mx = fmaxf(mx, __shfl_xor(mx, d, 32));
            const float mnew = fmaxf(mrun[r], mx);
            const float scl  = __expf(mrun[r] - mnew);
            const float p0 = __expf(v0 - mnew);
            const float p1 = __expf(v1 - mnew);
            float rs = p0 + p1;
#pragma unroll
            for (int d = 1; d < 16; d <<= 1) rs += __shfl_xor(rs, d, 32);
            lrun[r] = lrun[r] * scl + rs;
            mrun[r] = mnew;
#pragma unroll
            for (int t = 0; t < 4; ++t) O[t][r] *= scl;
            myP[(r + half * 8) * PROW + m16]      = f32_bf16(p0);
            myP[(r + half * 8) * PROW + 16 + m16] = f32_bf16(p1);
        }
        // Re-layout P (C-frag -> A-frag) through LDS.
        const ushort_t* prow = myP + m16 * PROW;
        Frag16 pf;
        pf.q[0] = *(const u32x4*)(prow + half * 8);
        pf.q[1] = *(const u32x4*)(prow + 16 + half * 8);
        // O += P(16x32) @ V(32x64); Vt rows are contiguous B-fragments.
#pragma unroll
        for (int t = 0; t < 4; ++t) {
            const ushort_t* vr = V + (size_t)(t * 16 + m16) * L_ + c0;
            v16bf bv = load_frag(vr + half * 8, vr + half * 8 + 16);
            O[t] = wmma_bf16(pf.v, bv, O[t]);
        }
    }

#pragma unroll
    for (int r = 0; r < 8; ++r) {
        const float invl = 1.0f / lrun[r];
        const int l = q0row + half * 8 + r;
        const size_t o = ((size_t)(b * L_ + l)) * EMB + h * HEAD;
#pragma unroll
        for (int t = 0; t < 4; ++t)
            AO[o + t * 16 + m16] = f32_bf16(O[t][r] * invl);
    }
}

// ---------------- Kernel 3: output projection + bias ----------------
__global__ void __launch_bounds__(256) proj(
    const ushort_t* __restrict__ AO, const ushort_t* __restrict__ Wpt,
    const float* __restrict__ bp, float* __restrict__ out) {

    const int lane = threadIdx.x & 31;
    const int wave = threadIdx.x >> 5;
    const int gw   = blockIdx.x * 8 + wave;    // 0..6143
    const int rt   = gw / HEAD_NUM;            // 512 row tiles
    const int ct   = gw % HEAD_NUM;            // 12 col groups of 64
    const int row0 = rt * 16;
    const int col0 = ct * 64;
    const int half = lane >> 4;
    const int m16  = lane & 15;

    v8f acc[4];
#pragma unroll
    for (int t = 0; t < 4; ++t) acc[t] = zero8();

    const ushort_t* arow = AO + (size_t)(row0 + m16) * EMB;

#pragma unroll 1
    for (int ks = 0; ks < EMB / 32; ++ks) {
        const int k0 = ks * 32 + half * 8;
        v16bf a = load_frag(arow + k0, arow + k0 + 16);
#pragma unroll
        for (int t = 0; t < 4; ++t) {
            const size_t wro = (size_t)(col0 + t * 16 + m16) * EMB + k0;
            v16bf bfr = load_frag(Wpt + wro, Wpt + wro + 16);
            acc[t] = wmma_bf16(a, bfr, acc[t]);
        }
    }

#pragma unroll
    for (int t = 0; t < 4; ++t) {
        const float bias = bp[col0 + t * 16 + m16];
#pragma unroll
        for (int r = 0; r < 8; ++r)
            out[(size_t)(row0 + half * 8 + r) * EMB + col0 + t * 16 + m16] = acc[t][r] + bias;
    }
}

// ---------------- Host launcher ----------------
extern "C" void kernel_launch(void* const* d_in, const int* in_sizes, int n_in,
                              void* d_out, int out_size, void* d_ws, size_t ws_size,
                              hipStream_t stream) {
    const float* emb = (const float*)d_in[0];
    const int*   pos = (const int*)  d_in[1];
    const float* Wq  = (const float*)d_in[2];
    const float* Wk  = (const float*)d_in[3];
    const float* Wv  = (const float*)d_in[4];
    const float* Wp  = (const float*)d_in[5];
    const float* bp  = (const float*)d_in[6];
    float* out = (float*)d_out;

    char* ws = (char*)d_ws;
    size_t off = 0;
    auto alloc = [&](size_t bytes) -> void* {
        void* p = ws + off;
        off += (bytes + 255) & ~(size_t)255;
        return p;
    };
    ushort_t* embb = (ushort_t*)alloc((size_t)ROWS * EMB * 2);
    ushort_t* Wqt  = (ushort_t*)alloc((size_t)EMB * EMB * 2);
    ushort_t* Wkt  = (ushort_t*)alloc((size_t)EMB * EMB * 2);
    ushort_t* Wvt  = (ushort_t*)alloc((size_t)EMB * EMB * 2);
    ushort_t* Wpt  = (ushort_t*)alloc((size_t)EMB * EMB * 2);
    ushort_t* Qr   = (ushort_t*)alloc((size_t)B_ * HEAD_NUM * L_ * HEAD * 2);
    ushort_t* Kr   = (ushort_t*)alloc((size_t)B_ * HEAD_NUM * L_ * HEAD * 2);
    ushort_t* Vt   = (ushort_t*)alloc((size_t)B_ * HEAD_NUM * HEAD * L_ * 2);
    ushort_t* AO   = (ushort_t*)alloc((size_t)ROWS * EMB * 2);

    const int nEmb = ROWS * EMB;                          // 6,291,456
    cvt_bf16<<<nEmb / (256 * 4), 256, 0, stream>>>(emb, embb, nEmb);
    const int nW = EMB * EMB;                             // 589,824
    xpose_w<<<nW / 256, 256, 0, stream>>>(Wq, Wqt);
    xpose_w<<<nW / 256, 256, 0, stream>>>(Wk, Wkt);
    xpose_w<<<nW / 256, 256, 0, stream>>>(Wv, Wvt);
    xpose_w<<<nW / 256, 256, 0, stream>>>(Wp, Wpt);

    qkv_rope<<<(ROWS / 16) * HEAD_NUM / 8, 256, 0, stream>>>(
        embb, Wqt, Wkt, Wvt, pos, Qr, Kr, Vt);            // 768 blocks
    attn<<<B_ * HEAD_NUM * 8, 256, 0, stream>>>(Qr, Kr, Vt, AO);   // 768 blocks
    proj<<<(ROWS / 16) * HEAD_NUM / 8, 256, 0, stream>>>(AO, Wpt, bp, out); // 768 blocks
}
